// ReadoutModule_72292889526462
// MI455X (gfx1250) — compile-verified, matrix operand order checked by reading, not run on hardware
//
#include <hip/hip_runtime.h>
#include <hip/hip_bf16.h>

#define N_NODES 100000
#define E_EDGES 1600000
#define NS_IN   128
#define MUL_IN  32
#define NS_OUT  64
#define MUL_OUT 16
#define HID     256
#define FEAT_W  224   // NS_IN + 3*MUL_IN
#define OUT_W   112   // NS_OUT + 3*MUL_OUT
#define TM      64    // rows (nodes) per block

// padded LDS row strides (bf16 elements): +8 rotates banks, keeps 16B alignment
#define A_STRIDE 136  // 128 + 8  -> 272 B rows
#define H_STRIDE 520  // 512 + 8  -> 1040 B rows

typedef __attribute__((ext_vector_type(16))) __bf16 v16bf;
typedef __attribute__((ext_vector_type(8)))  float  v8f;

// ---- workspace layout (bytes, all 32B aligned) ----
#define WS_MASK 0                  // u32[N]
#define WS_PW1I 409600             // packed inv_W1  (128x256) bf16 : 131072 B
#define WS_PW1W (WS_PW1I+131072)   // packed wemb_W1 (128x256) bf16 : 131072 B
#define WS_PW2I (WS_PW1W+131072)   // packed inv_W2  (256x64)  bf16 : 32768 B
#define WS_PW2W (WS_PW2I+32768)    // packed wemb_W2 (256x512) bf16 : 262144 B

__global__ void k_zero_mask(unsigned* __restrict__ mask) {
    int i = blockIdx.x * 256 + threadIdx.x;
    if (i < N_NODES) mask[i] = 0u;
}

__global__ void k_scatter_mask(const int* __restrict__ edge_src,
                               unsigned* __restrict__ mask) {
    int i = blockIdx.x * 256 + threadIdx.x;
    if (i < E_EDGES) mask[edge_src[i]] = 1u;   // benign races
}

// K index within a 32-wide K block for A/B fragment element e on this lane
__device__ __forceinline__ int frag_k(int lane, int e) {
    return ((e >> 3) << 4) + ((lane >> 4) << 3) + (e & 7);
}

// Pack a row-major f32 weight (K x Ncols) into B-fragment-major bf16:
// fragment id = nt*KT + kt ; per lane 16 contiguous bf16 (32 bytes).
__global__ void k_pack_w(const float* __restrict__ W, __bf16* __restrict__ dst,
                         int K, int Ncols) {
    int KT = K >> 5, NT = Ncols >> 4;
    int gid  = blockIdx.x * 256 + threadIdx.x;
    int fid  = gid >> 5;
    if (fid >= KT * NT) return;
    int lane = gid & 31;
    int nt = fid / KT, kt = fid % KT;
    int col = (nt << 4) + (lane & 15);
    __bf16* o = dst + ((size_t)fid * 32 + lane) * 16;
#pragma unroll
    for (int e = 0; e < 16; ++e) {
        int k = (kt << 5) + frag_k(lane, e);
        o[e] = (__bf16)W[(size_t)k * Ncols + col];
    }
}

// Load an A fragment (16x32 bf16) from row-major bf16 LDS tile.
// Elements e=0..7  = K base..base+7     (16 contiguous bytes)
// Elements e=8..15 = K base+16..base+23 (16 contiguous bytes)
// -> exactly two ds_load_b128 per lane.
__device__ __forceinline__ v16bf load_a_frag(const __bf16* lds, int stride,
                                             int m0, int k0, int lane) {
    union { v16bf v; uint4 q[2]; } a;
    int row  = m0 + (lane & 15);
    int base = k0 + ((lane >> 4) << 3);
    const char* p = (const char*)lds + ((size_t)row * stride + base) * 2;
    a.q[0] = *(const uint4*)p;
    a.q[1] = *(const uint4*)(p + 32);
    return a.v;
}

__device__ __forceinline__ float silu_fast(float v) {
    // v * rcp(1 + exp(-v)) : v_exp_f32 + v_rcp_f32, no IEEE div expansion
    return v * __builtin_amdgcn_rcpf(1.0f + __expf(-v));
}

__global__ __launch_bounds__(256)
void k_fused(const float* __restrict__ feat,
             const __bf16* __restrict__ pw1i, const __bf16* __restrict__ pw1w,
             const __bf16* __restrict__ pw2i, const __bf16* __restrict__ pw2w,
             const unsigned* __restrict__ mask,
             float* __restrict__ out) {
    __shared__ __bf16 Hs[TM * H_STRIDE];   // hidden [h_inv | h_wemb] (st.2-3), x stage (st.4)
    __shared__ __bf16 Buf[TM * H_STRIDE];  // stage2 A-tile / stage3 w-output
    __shared__ float  maskf[TM];

    const int tid  = threadIdx.x;
    const int lane = tid & 31;
    const int wave = tid >> 5;
    const int row0 = blockIdx.x * TM;

    // ---------- stage 1: scalars (64x128 f32) -> bf16 A tile in Buf ----------
    {
        int r  = tid >> 2;
        int c0 = (tid & 3) << 5;
        int n  = row0 + r;
        if (n < N_NODES) {
#pragma unroll
            for (int j = 0; j < 8; ++j) {
                float4 v = *(const float4*)(feat + (size_t)n * FEAT_W + c0 + j * 4);
                __bf16* d = &Buf[r * A_STRIDE + c0 + j * 4];
                d[0] = (__bf16)v.x; d[1] = (__bf16)v.y;
                d[2] = (__bf16)v.z; d[3] = (__bf16)v.w;
            }
        } else {
#pragma unroll
            for (int j = 0; j < 32; ++j) Buf[r * A_STRIDE + c0 + j] = (__bf16)0.0f;
        }
        if (tid < TM) {
            int nn = row0 + tid;
            maskf[tid] = (nn < N_NODES && mask[nn]) ? 1.0f : 0.0f;
        }
    }
    __syncthreads();

    // ---------- stage 2: hidden = silu(A @ [inv_W1 | wemb_W1] / sqrt(128)) ----------
    const float s1 = 0.08838834764831845f;   // 1/sqrt(128)
    for (int t = wave; t < 128; t += 8) {    // tile = ntc*4 + mt (uniform per wave)
        int ntc = t >> 2, mt = t & 3;
        const __bf16* pw = (ntc < 16) ? pw1i : pw1w;
        int nt = (ntc < 16) ? ntc : ntc - 16;
        v8f c = {};
#pragma unroll
        for (int kt = 0; kt < 4; ++kt) {
            v16bf a = load_a_frag(Buf, A_STRIDE, mt * 16, kt * 32, lane);
            v16bf b = *(const v16bf*)(pw + ((size_t)(nt * 4 + kt) * 32 + lane) * 16);
            c = __builtin_amdgcn_wmma_f32_16x16x32_bf16(false, a, false, b,
                                                        (short)0, c, false, false);
        }
        int mloc = mt * 16 + ((lane >> 4) << 3);
        int col  = ntc * 16 + (lane & 15);
#pragma unroll
        for (int r = 0; r < 8; ++r)
            Hs[(mloc + r) * H_STRIDE + col] = (__bf16)silu_fast(c[r] * s1);
    }
    __syncthreads();

    // ---------- stage 3: inv_out = h_inv @ inv_W2 /16 ; w = h_wemb @ wemb_W2 /16 ----------
    const float s2 = 0.0625f;                 // 1/sqrt(256)
    for (int t = wave; t < 144; t += 8) {     // 16 inv tiles + 128 wemb tiles
        bool is_inv = (t < 16);
        int tt = is_inv ? t : t - 16;
        int nt = tt >> 2, mt = tt & 3;
        const __bf16* pw = is_inv ? pw2i : pw2w;
        int koff = is_inv ? 0 : 256;
        v8f c = {};
#pragma unroll
        for (int kt = 0; kt < 8; ++kt) {
            v16bf a = load_a_frag(Hs, H_STRIDE, mt * 16, koff + kt * 32, lane);
            v16bf b = *(const v16bf*)(pw + ((size_t)(nt * 8 + kt) * 32 + lane) * 16);
            c = __builtin_amdgcn_wmma_f32_16x16x32_bf16(false, a, false, b,
                                                        (short)0, c, false, false);
        }
        int mloc = mt * 16 + ((lane >> 4) << 3);
        int col  = nt * 16 + (lane & 15);
        if (is_inv) {
#pragma unroll
            for (int r = 0; r < 8; ++r) {
                int lr = mloc + r;
                int n  = row0 + lr;
                if (n < N_NODES)
                    out[(size_t)n * OUT_W + col] = c[r] * s2 * maskf[lr];
            }
        } else {
#pragma unroll
            for (int r = 0; r < 8; ++r)
                Buf[(mloc + r) * H_STRIDE + col] = (__bf16)(c[r] * s2);
        }
    }
    __syncthreads();

    // ---------- stage 4a: stage x_eq (64 x 96 f32) into Hs (reused) ----------
    {
        float* xs = (float*)Hs;
        int r  = tid >> 2;
        int j0 = (tid & 3) * 24;
        int n  = row0 + r;
#pragma unroll
        for (int j = 0; j < 24; ++j)
            xs[r * 96 + j0 + j] =
                (n < N_NODES) ? feat[(size_t)n * FEAT_W + NS_IN + j0 + j] : 0.0f;
    }
    __syncthreads();

    // ---------- stage 4b: eq_out[o,m] = sum_i x[i,m]*w[i,o] / sqrt(32), masked ----------
    {
        int r = tid >> 2, part = tid & 3;
        int n = row0 + r;
        if (n < N_NODES) {
            const float rs = 0.17677669529663687f;   // 1/sqrt(32)
            float mval = maskf[r];
            const float* xr = (const float*)Hs + r * 96;
            float acc[4][3];
#pragma unroll
            for (int q = 0; q < 4; ++q)
                acc[q][0] = acc[q][1] = acc[q][2] = 0.0f;
            for (int i = 0; i < MUL_IN; ++i) {
                float x0 = xr[i * 3 + 0], x1 = xr[i * 3 + 1], x2 = xr[i * 3 + 2];
#pragma unroll
                for (int q = 0; q < 4; ++q) {
                    float wv = (float)Buf[r * H_STRIDE + i * 16 + part * 4 + q];
                    acc[q][0] += x0 * wv; acc[q][1] += x1 * wv; acc[q][2] += x2 * wv;
                }
            }
#pragma unroll
            for (int q = 0; q < 4; ++q) {
                int o = part * 4 + q;
#pragma unroll
                for (int m = 0; m < 3; ++m)
                    out[(size_t)n * OUT_W + NS_OUT + o * 3 + m] = acc[q][m] * rs * mval;
            }
        }
    }
}

extern "C" void kernel_launch(void* const* d_in, const int* in_sizes, int n_in,
                              void* d_out, int out_size, void* d_ws, size_t ws_size,
                              hipStream_t stream) {
    const float* feat   = (const float*)d_in[0];
    const int*   edge   = (const int*)d_in[1];   // (2,E): first E = sources
    const float* invW1  = (const float*)d_in[2];
    const float* invW2  = (const float*)d_in[3];
    const float* wembW1 = (const float*)d_in[4];
    const float* wembW2 = (const float*)d_in[5];

    char* ws = (char*)d_ws;
    unsigned* mask = (unsigned*)(ws + WS_MASK);
    __bf16* pw1i = (__bf16*)(ws + WS_PW1I);
    __bf16* pw1w = (__bf16*)(ws + WS_PW1W);
    __bf16* pw2i = (__bf16*)(ws + WS_PW2I);
    __bf16* pw2w = (__bf16*)(ws + WS_PW2W);

    k_zero_mask<<<(N_NODES + 255) / 256, 256, 0, stream>>>(mask);
    k_scatter_mask<<<(E_EDGES + 255) / 256, 256, 0, stream>>>(edge, mask);

    // fragment counts: W1: 4*16=64 tiles ; inv_W2: 8*4=32 ; wemb_W2: 8*32=256
    k_pack_w<<<(64  * 32 + 255) / 256, 256, 0, stream>>>(invW1,  pw1i, 128, 256);
    k_pack_w<<<(64  * 32 + 255) / 256, 256, 0, stream>>>(wembW1, pw1w, 128, 256);
    k_pack_w<<<(32  * 32 + 255) / 256, 256, 0, stream>>>(invW2,  pw2i, 256, 64);
    k_pack_w<<<(256 * 32 + 255) / 256, 256, 0, stream>>>(wembW2, pw2w, 256, 512);

    k_fused<<<(N_NODES + TM - 1) / TM, 256, 0, stream>>>(
        feat, pw1i, pw1w, pw2i, pw2w, mask, (float*)d_out);
}
